// SelfAttentionEdgeIndexCreatorLayer_52055003627921
// MI455X (gfx1250) — matrix operand order repaired and checked by reading, not compiled, not run on hardware
//
#include <hip/hip_runtime.h>
#include <hip/hip_bf16.h>

typedef __attribute__((ext_vector_type(16))) _Float16 v16h;
typedef __attribute__((ext_vector_type(8)))  _Float16 v8h;
typedef __attribute__((ext_vector_type(8)))  float    v8f;

#define EMBED    256
#define HEADS    8
#define HEAD_DIM 32
#define SEQ      2048
#define BATCH    8
#define N_EDGES  32
#define KTILES   (SEQ / 16)   // 128

__device__ __forceinline__ v16h cat8(v8h lo, v8h hi) {
  return __builtin_shufflevector(lo, hi, 0,1,2,3,4,5,6,7,8,9,10,11,12,13,14,15);
}

// ---------------------------------------------------------------------------
// Kernel 1: per-head linear projection  y = x @ W^T  (rows = B*L*H, K=N=32)
// One wave handles a 16-row tile with two v_wmma_f32_16x16x32_f16 (N halves).
// Output stored as f16 (feeds WMMA B-fragments directly, halves L2 traffic).
// ---------------------------------------------------------------------------
__global__ void __launch_bounds__(256)
proj_f16_kernel(const float* __restrict__ X,
                const float* __restrict__ W,
                _Float16* __restrict__ Y,
                int ntiles) {
  int gwave = (int)((blockIdx.x * blockDim.x + threadIdx.x) >> 5);
  if (gwave >= ntiles) return;              // uniform per wave: EXEC stays all-ones
  int lane = threadIdx.x & 31;
  int nl = lane & 15;                       // A: row m   | B: col n | D: col n
  int hb = lane >> 4;                       // A: k chunk | B: k half| D: m group

  // A fragment (16x32 f16): lane holds K = 16*hb + {0..15}
  const float* arow = X + ((size_t)gwave * 16 + nl) * HEAD_DIM;
  v16h a;
  #pragma unroll
  for (int j = 0; j < 8; ++j) {
    a[j]     = (_Float16)arow[hb * 16 + j];
    a[8 + j] = (_Float16)arow[hb * 16 + 8 + j];
  }
  // B fragments: B[d][e] = W[e][d]; lane covers d = 16*hb .. +15 for column e
  const float* w0 = W + (size_t)nl * HEAD_DIM + hb * 16;         // e = nl
  const float* w1 = W + (size_t)(nl + 16) * HEAD_DIM + hb * 16;  // e = nl + 16
  v16h b0, b1;
  #pragma unroll
  for (int j = 0; j < 16; ++j) { b0[j] = (_Float16)w0[j]; b1[j] = (_Float16)w1[j]; }

  v8f c0 = {}, c1 = {};
  c0 = __builtin_amdgcn_wmma_f32_16x16x32_f16(false, a, false, b0, (short)0, c0, false, false);
  c1 = __builtin_amdgcn_wmma_f32_16x16x32_f16(false, a, false, b1, (short)0, c1, false, false);

  #pragma unroll
  for (int r = 0; r < 8; ++r) {
    size_t row = (size_t)gwave * 16 + hb * 8 + r;   // D: VGPR r -> M = r + 8*hb
    Y[row * HEAD_DIM + nl]      = (_Float16)c0[r];
    Y[row * HEAD_DIM + 16 + nl] = (_Float16)c1[r];
  }
}

// ---------------------------------------------------------------------------
// Kernel 2: fused attention scores -> per-head softmax -> head mean -> top-32.
// One 128-thread workgroup (4 waves) per (batch, 16-query tile).
// LDS: 16x2048 f32 mean-attn accumulator (128KB) + 8KB Q' tile + stats.
// Score tile = one v_wmma_f32_16x16x32_f16 per (head, key-tile).
// ---------------------------------------------------------------------------
__global__ void __launch_bounds__(128)
attn_topk_kernel(const _Float16* __restrict__ Qp,
                 const _Float16* __restrict__ Kp,
                 float* __restrict__ out) {
  extern __shared__ char smem[];
  float*    S  = (float*)smem;                                    // [16][SEQ]
  _Float16* Qt = (_Float16*)(smem + 16 * SEQ * 4);                // [16][EMBED]
  float*    Mh = (float*)(smem + 16 * SEQ * 4 + 16 * EMBED * 2);  // [H][16] row max
  float*    Lh = Mh + HEADS * 16;                                 // [H][16] 1/sumexp
  float*    Pm = Lh + HEADS * 16;                                 // [4][16] wave partials
  float*    Pl = Pm + 4 * 16;

  const int tid  = threadIdx.x;
  const int wave = tid >> 5;
  const int lane = tid & 31;
  const int nl = lane & 15;   // A row / B col / D col
  const int hb = lane >> 4;   // A chunk / B k-half / D m-group

  const int b  = blockIdx.x >> 7;
  const int qt = blockIdx.x & 127;

  // ---- stage Q' tile into LDS (16 rows x 256 halves, contiguous) ----
  {
    const uint4* src = (const uint4*)(Qp + ((size_t)b * SEQ + (size_t)qt * 16) * EMBED);
    uint4* dst = (uint4*)Qt;
    for (int i = tid; i < (16 * EMBED * 2) / 16; i += 128) dst[i] = src[i];
  }
  __syncthreads();

  const float  scale = 0.0625f;            // 1/sqrt(EMBED)
  const size_t krow0 = (size_t)b * SEQ;

  // ---- sweep 1: per-head online softmax stats (row max, sum of exp) ----
  for (int h = 0; h < HEADS; ++h) {
    const _Float16* ap = Qt + (size_t)nl * EMBED + h * HEAD_DIM;
    v16h a = cat8(*(const v8h*)(ap + hb * 16), *(const v8h*)(ap + hb * 16 + 8));

    float mr[8], lr[8];
    #pragma unroll
    for (int r = 0; r < 8; ++r) { mr[r] = -1e30f; lr[r] = 0.0f; }

    for (int kt = wave; kt < KTILES; kt += 4) {
      const _Float16* bp =
          Kp + ((krow0 + (size_t)kt * 16 + nl) * HEADS + h) * HEAD_DIM + hb * 16;
      if (kt + 4 < KTILES) __builtin_prefetch(bp + (size_t)64 * EMBED, 0, 1);
      v16h bf = *(const v16h*)bp;           // 16 contiguous f16 = B fragment
      v8f c = {};
      c = __builtin_amdgcn_wmma_f32_16x16x32_f16(false, a, false, bf, (short)0, c, false, false);
      #pragma unroll
      for (int r = 0; r < 8; ++r) {
        float e  = c[r] * scale;
        float nm = fmaxf(mr[r], e);
        lr[r] = lr[r] * __expf(mr[r] - nm) + __expf(e - nm);
        mr[r] = nm;
      }
    }
    // reduce (m,l) across the 16 lanes of each half-wave (N direction)
    #pragma unroll
    for (int r = 0; r < 8; ++r) {
      float m = mr[r], l = lr[r];
      #pragma unroll
      for (int off = 1; off < 16; off <<= 1) {
        float om = __shfl_xor(m, off, 32);
        float ol = __shfl_xor(l, off, 32);
        float nm = fmaxf(m, om);
        l = l * __expf(m - nm) + ol * __expf(om - nm);
        m = nm;
      }
      mr[r] = m; lr[r] = l;
    }
    if (nl == 0) {                           // lanes 0 and 16 publish 8 rows each
      #pragma unroll
      for (int r = 0; r < 8; ++r) {
        Pm[wave * 16 + hb * 8 + r] = mr[r];
        Pl[wave * 16 + hb * 8 + r] = lr[r];
      }
    }
    __syncthreads();
    if (tid < 16) {                          // merge 4 wave partials per row
      float m = Pm[tid], l = Pl[tid];
      #pragma unroll
      for (int w2 = 1; w2 < 4; ++w2) {
        float om = Pm[w2 * 16 + tid], ol = Pl[w2 * 16 + tid];
        float nm = fmaxf(m, om);
        l = l * __expf(m - nm) + ol * __expf(om - nm);
        m = nm;
      }
      Mh[h * 16 + tid] = m;
      Lh[h * 16 + tid] = 1.0f / l;
    }
    __syncthreads();
  }

  // ---- zero mean-attention accumulator ----
  for (int i = tid; i < 16 * SEQ; i += 128) S[i] = 0.0f;
  __syncthreads();

  // ---- sweep 2: accumulate head-mean attention into S ----
  // Each (row,col) has a fixed owner (wave,lane) across all heads: no races.
  for (int h = 0; h < HEADS; ++h) {
    const _Float16* ap = Qt + (size_t)nl * EMBED + h * HEAD_DIM;
    v16h a = cat8(*(const v8h*)(ap + hb * 16), *(const v8h*)(ap + hb * 16 + 8));
    float mrow[8], irow[8];
    #pragma unroll
    for (int r = 0; r < 8; ++r) {
      mrow[r] = Mh[h * 16 + hb * 8 + r];
      irow[r] = Lh[h * 16 + hb * 8 + r] * 0.125f;   // fold 1/HEADS
    }
    for (int kt = wave; kt < KTILES; kt += 4) {
      const _Float16* bp =
          Kp + ((krow0 + (size_t)kt * 16 + nl) * HEADS + h) * HEAD_DIM + hb * 16;
      if (kt + 4 < KTILES) __builtin_prefetch(bp + (size_t)64 * EMBED, 0, 1);
      v16h bf = *(const v16h*)bp;
      v8f c = {};
      c = __builtin_amdgcn_wmma_f32_16x16x32_f16(false, a, false, bf, (short)0, c, false, false);
      #pragma unroll
      for (int r = 0; r < 8; ++r) {
        int row = hb * 8 + r;
        float p = __expf(c[r] * scale - mrow[r]) * irow[r];
        S[row * SEQ + kt * 16 + nl] += p;
      }
    }
  }
  __syncthreads();

  // ---- top-32 per row (iterative wave argmax, low-index tie-break), then
  //      rank-sort the 32 winners ascending and emit [B,2,Lq*N_EDGES] ----
  const size_t obase = (size_t)b * 2 * SEQ * N_EDGES;
  for (int rr = wave; rr < 16; rr += 4) {
    float* row = S + (size_t)rr * SEQ;
    int myidx = 0;
    for (int e = 0; e < N_EDGES; ++e) {
      float bv = -1e30f; int bi = 0;
      #pragma unroll 4
      for (int j = 0; j < SEQ / 32; ++j) {
        int cidx = lane + 32 * j;            // column owner == lane: safe clears
        float v = row[cidx];
        if (v > bv || (v == bv && cidx < bi)) { bv = v; bi = cidx; }
      }
      #pragma unroll
      for (int off = 16; off >= 1; off >>= 1) {
        float ov = __shfl_xor(bv, off, 32);
        int   oi = __shfl_xor(bi, off, 32);
        if (ov > bv || (ov == bv && oi < bi)) { bv = ov; bi = oi; }
      }
      if (lane == e) myidx = bi;             // lane e keeps e-th largest
      if (lane == (bi & 31)) row[bi] = -1e30f;  // only owner lane re-reads it
    }
    int rank = 0;                            // indices are distinct: exact ranks
    #pragma unroll
    for (int j = 0; j < 32; ++j) {
      int oj = __shfl(myidx, j, 32);
      rank += (oj < myidx) ? 1 : 0;
    }
    int q = qt * 16 + rr;
    out[obase + (size_t)q * N_EDGES + lane] = (float)q;                          // row 0: node
    out[obase + (size_t)SEQ * N_EDGES + (size_t)q * N_EDGES + rank] = (float)myidx; // row 1: neighbor
  }
}

// ---------------------------------------------------------------------------
extern "C" void kernel_launch(void* const* d_in, const int* in_sizes, int n_in,
                              void* d_out, int out_size, void* d_ws, size_t ws_size,
                              hipStream_t stream) {
  const float* keys  = (const float*)d_in[0];   // [B, L, 256]
  const float* query = (const float*)d_in[1];   // [B, L, 256]
  const float* Wk    = (const float*)d_in[2];   // [32, 32]
  const float* Wq    = (const float*)d_in[3];   // [32, 32]

  _Float16* Kp = (_Float16*)d_ws;                       // 8 MB f16 K'
  _Float16* Qp = Kp + (size_t)BATCH * SEQ * EMBED;      // 8 MB f16 Q'

  const int rows   = BATCH * SEQ * HEADS;   // 131072 projection rows
  const int ntiles = rows / 16;             // 8192 wave tiles
  const int blocks = ntiles / 8;            // 8 waves per 256-thread block

  proj_f16_kernel<<<blocks, 256, 0, stream>>>(keys,  Wk, Kp, ntiles);
  proj_f16_kernel<<<blocks, 256, 0, stream>>>(query, Wq, Qp, ntiles);

  const size_t smem = (size_t)16 * SEQ * 4        // mean-attn accumulator
                    + (size_t)16 * EMBED * 2      // Q' tile
                    + (size_t)(HEADS * 16 * 2 + 4 * 16 * 2) * 4;  // softmax stats
  attn_topk_kernel<<<BATCH * KTILES, 128, smem, stream>>>(Qp, Kp, (float*)d_out);

  (void)in_sizes; (void)n_in; (void)out_size; (void)ws_size;
}
